// Attention_68822555951548
// MI455X (gfx1250) — matrix operand order repaired and checked by reading, not compiled, not run on hardware
//
#include <hip/hip_runtime.h>
#include <hip/hip_bf16.h>

#define HEADS 8
#define DK 64
#define SEQ 2048
#define DMODEL 512

typedef __attribute__((ext_vector_type(16))) _Float16 v16h;
typedef __attribute__((ext_vector_type(8)))  float    v8f;
typedef __attribute__((ext_vector_type(4)))  unsigned int v4u;
typedef __attribute__((ext_vector_type(8)))  int         v8i;
typedef __attribute__((ext_vector_type(4)))  int         v4i;

#if __has_builtin(__builtin_amdgcn_tensor_load_to_lds) && __has_builtin(__builtin_amdgcn_s_wait_tensorcnt)
#define USE_TDM 1
#else
#define USE_TDM 0
#endif

// POD 16-byte chunks (HIP's uint4/float4 have non-trivial ctors -> unsafe in unions)
struct alignas(16) U128 { unsigned int x, y, z, w; };
struct alignas(16) F4   { float x, y, z, w; };

union FragU { v16h h; U128 u[2]; };

// A-matrix fragment (16x32 f16): lane holds row (lane&15); half j maps to
// k = j + 8*(j>=8) + 8*(lane>=16)  ->  two contiguous 16B reads at k=8*grp and 16+8*grp.
__device__ __forceinline__ v16h load_frag_a(const _Float16* base, int ld, int lane) {
  const int row = lane & 15, grp = lane >> 4;
  const _Float16* p = base + (size_t)row * ld + 8 * grp;
  FragU f;
  f.u[0] = *reinterpret_cast<const U128*>(p);
  f.u[1] = *reinterpret_cast<const U128*>(p + 16);
  return f.h;
}

// B-matrix fragment (32x16 f16), staged as [n][k] (k contiguous): lane holds col (lane&15);
// half j maps to k = j + 8*(j>=8) + 8*(lane>=16) -> two contiguous 16B reads at k=8*grp and 16+8*grp.
__device__ __forceinline__ v16h load_frag_b(const _Float16* base, int ld, int lane) {
  const int row = lane & 15, grp = lane >> 4;
  const _Float16* p = base + (size_t)row * ld + 8 * grp;
  FragU f;
  f.u[0] = *reinterpret_cast<const U128*>(p);
  f.u[1] = *reinterpret_cast<const U128*>(p + 16);
  return f.h;
}

// ---- DPP16 row-rotate butterfly reduction (VALU; no LDS-pipe traffic) ----
// row_ror:N control = 0x120|N; rotations by 1,2,4,8 within each 16-lane row
// give every lane the full 16-lane reduction.
template <int CTRL>
__device__ __forceinline__ float dpp_rot(float x) {
  const int xi = __float_as_int(x);
  return __int_as_float(__builtin_amdgcn_update_dpp(xi, xi, CTRL, 0xF, 0xF, true));
}
__device__ __forceinline__ float rowmax16(float x) {
  x = fmaxf(x, dpp_rot<0x121>(x));
  x = fmaxf(x, dpp_rot<0x122>(x));
  x = fmaxf(x, dpp_rot<0x124>(x));
  x = fmaxf(x, dpp_rot<0x128>(x));
  return x;
}

#if USE_TDM
// ---- Tensor Data Mover descriptor packing (CDNA5 ISA) ----
// Group 0: [1:0]=count=1, [63:32]=lds_addr, [120:64]=global_addr, [127:126]=type=2
__device__ __forceinline__ v4u tdm_group0(unsigned lds_off, unsigned long long gaddr) {
  v4u g;
  g.x = 1u;
  g.y = lds_off;
  g.z = (unsigned)(gaddr & 0xFFFFFFFFu);
  g.w = (unsigned)((gaddr >> 32) & 0x01FFFFFFu) | (2u << 30);
  return g;
}

// Group 1 for a 2D tile with LDS padding:
// [17:16]=data_size, [20]=pad_enable, [24:22]=pad_interval, [31:25]=pad_amount,
// [79:48]=tensor_dim0, [111:80]=tensor_dim1, [127:112]=tile_dim0, [143:128]=tile_dim1,
// [207:160]=tensor_dim0_stride. atomic-barrier/iterate/multicast all 0.
__device__ __forceinline__ v8i tdm_group1_2d(unsigned ds_code, unsigned pad_int,
                                             unsigned pad_amt, unsigned tdim0,
                                             unsigned tdim1, unsigned tile0,
                                             unsigned tile1,
                                             unsigned long long stride0) {
  v8i g;
  g[0] = (int)((ds_code << 16) | (1u << 20) | (pad_int << 22) | (pad_amt << 25));
  g[1] = (int)((tdim0 & 0xFFFFu) << 16);
  g[2] = (int)(((tdim0 >> 16) & 0xFFFFu) | ((tdim1 & 0xFFFFu) << 16));
  g[3] = (int)(((tdim1 >> 16) & 0xFFFFu) | ((tile0 & 0xFFFFu) << 16));
  g[4] = (int)(tile1 & 0xFFFFu);                 // tile_dim2 = 0
  g[5] = (int)(stride0 & 0xFFFFFFFFull);
  g[6] = (int)((stride0 >> 32) & 0xFFFFull);     // tensor_dim1_stride = 0 (2D)
  g[7] = 0;
  return g;
}
#endif

// ---------------------------------------------------------------------------
// Projection GEMM: Y(8192x512) = X(8192x512,f32) @ W(512x512,f32), f16 out.
// Block = 256 thr (8 waves), tile 128x64, K-step 32.
// MODE 0/1: Y stored (B,H,S,dk); MODE 2: Y stored transposed (B,H,dk,S).
// ---------------------------------------------------------------------------
template <int MODE>
__global__ __launch_bounds__(256)
void proj_gemm_kernel(const float* __restrict__ X, const float* __restrict__ W,
                      _Float16* __restrict__ Y, float scale) {
  __shared__ __align__(16) _Float16 As[128][40];  // [m][k], padded
  __shared__ __align__(16) _Float16 Bs[64][40];   // [n][k] = W^T tile, padded

  const int tid  = threadIdx.x;
  const int lane = tid & 31, wave = tid >> 5;
  const int M0 = blockIdx.x * 128, N0 = blockIdx.y * 64;
  const int grp = lane >> 4, col0 = lane & 15;

  v8f acc[4] = {};

  for (int k0 = 0; k0 < DMODEL; k0 += 32) {
    // stage A tile (128x32) with f32->f16 convert: 16 elems/thread
    {
      const int row = tid >> 1;
      const int cb  = (tid & 1) * 16;
      const float* src = X + (size_t)(M0 + row) * DMODEL + k0 + cb;
      _Float16* dst = &As[row][cb];
#pragma unroll
      for (int i = 0; i < 16; i += 4) {
        F4 f = *reinterpret_cast<const F4*>(src + i);
        dst[i + 0] = (_Float16)f.x; dst[i + 1] = (_Float16)f.y;
        dst[i + 2] = (_Float16)f.z; dst[i + 3] = (_Float16)f.w;
      }
    }
    // stage W^T tile (64n x 32k): 8 elems/thread, transposed store
    {
      const int kk = tid >> 3;
      const int nb = (tid & 7) * 8;
      const float* src = W + (size_t)(k0 + kk) * DMODEL + N0 + nb;
#pragma unroll
      for (int i = 0; i < 8; i += 4) {
        F4 f = *reinterpret_cast<const F4*>(src + i);
        Bs[nb + i + 0][kk] = (_Float16)f.x;
        Bs[nb + i + 1][kk] = (_Float16)f.y;
        Bs[nb + i + 2][kk] = (_Float16)f.z;
        Bs[nb + i + 3][kk] = (_Float16)f.w;
      }
    }
    __syncthreads();

    // batch all fragment loads, then issue the WMMA chain (pipelined ds loads)
    v16h a = load_frag_a(&As[wave * 16][0], 40, lane);
    v16h bfr[4];
#pragma unroll
    for (int g = 0; g < 4; ++g) bfr[g] = load_frag_b(&Bs[g * 16][0], 40, lane);
#pragma unroll
    for (int g = 0; g < 4; ++g)
      acc[g] = __builtin_amdgcn_wmma_f32_16x16x32_f16(false, a, false, bfr[g],
                                                      (short)0, acc[g], false, false);
    __syncthreads();
  }

  // store (C layout: VGPR r -> row r + 8*grp, col = lane&15); MODE is compile-time
#pragma unroll
  for (int g = 0; g < 4; ++g) {
#pragma unroll
    for (int r = 0; r < 8; ++r) {
      const int m    = M0 + wave * 16 + r + 8 * grp;
      const int bidx = m >> 11;           // m / SEQ
      const int s    = m & (SEQ - 1);
      const int col  = N0 + g * 16 + col0;
      const int h    = col >> 6;          // col / DK
      const int d    = col & (DK - 1);
      const _Float16 v = (_Float16)(acc[g][r] * scale);
      if (MODE == 2)
        Y[(((size_t)bidx * HEADS + h) * DK + d) * SEQ + s] = v;   // V^T layout
      else
        Y[(((size_t)bidx * HEADS + h) * SEQ + s) * DK + d] = v;
    }
  }
}

// ---------------------------------------------------------------------------
// Flash attention: grid (B*H, S/128), block 256 (8 waves x 16 query rows).
// Q frags in registers; 32-key tiles double-buffered in LDS via TDM DMA
// (tensor_load_to_lds, TENSORcnt), overlapped with WMMA; online softmax with
// DPP row-max and WMMA-ones row-sum.
// ---------------------------------------------------------------------------
__global__ __launch_bounds__(256)
void flash_attn_kernel(const _Float16* __restrict__ Q16,
                       const _Float16* __restrict__ K16,
                       const _Float16* __restrict__ VT16,
                       const int* __restrict__ V_len,
                       const int* __restrict__ Q_len,
                       float* __restrict__ Out) {
  __shared__ __align__(16) _Float16 Ks[2][32][72];    // [buf][key][dk], padded
  __shared__ __align__(16) _Float16 Vs[2][64][40];    // [buf][d][key] (V^T tile), padded
  __shared__ __align__(16) _Float16 Ps[8][16][40];    // per-wave P staging

  const int tid  = threadIdx.x;
  const int lane = tid & 31, wave = tid >> 5;
  const int bh = blockIdx.x;
  const int b  = bh / HEADS;
  const int h  = bh % HEADS;
  const int q0 = blockIdx.y * 128;
  const int grp = lane >> 4, col0 = lane & 15;

  // Q fragments (16 rows x 64 k) straight from global: row-contiguous
  const _Float16* qbase = Q16 + ((size_t)bh * SEQ + q0 + wave * 16) * DK;
  const v16h qa0 = load_frag_a(qbase, DK, lane);       // k = 0..31
  const v16h qa1 = load_frag_a(qbase + 32, DK, lane);  // k = 32..63

  // all-ones B matrix for WMMA row sums
  v16h ones;
#pragma unroll
  for (int i = 0; i < 16; ++i) ones[i] = (_Float16)1.0f;

  const int vlen = V_len[b];
  const int qlen = Q_len[b];
  int kmax = (vlen + 31) & ~31;
  if (kmax > SEQ) kmax = SEQ;
  const int ntiles = kmax >> 5;

  const _Float16* kgbase = K16 + (size_t)bh * SEQ * DK;
  const _Float16* vgbase = VT16 + (size_t)bh * DK * SEQ;

#if USE_TDM
  // K tile: 32 rows x 128B, contiguous in memory (stride = row length).
  // data_size=8B (code 3): tile 16x32 units; pad after every 16 units (code 4:
  // 1<<4 = 16 x 8B = 128B = one row) by 4 DWORDs (code 3) -> 72-half LDS rows.
  const v8i kG1 = tdm_group1_2d(3u, 4u, 3u, 16u, 32u, 16u, 32u, 16ull);
  // V^T tile: 64 rows x 64B, memory row stride = SEQ halves = 512 x 8B units.
  // pad after every 8 units (code 3) by 4 DWORDs (code 3) -> 40-half LDS rows.
  const v8i vG1 = tdm_group1_2d(3u, 3u, 3u, 8u, 64u, 8u, 64u, 512ull);
  const v4i z4 = {0, 0, 0, 0};
  const v8i z8 = {0, 0, 0, 0, 0, 0, 0, 0};

  // Prologue: issue tile 0 into buffer 0 (wave0 -> K, wave1 -> V^T).
  if (wave == 0) {
    v4u g0 = tdm_group0((unsigned)(uintptr_t)&Ks[0][0][0],
                        (unsigned long long)(uintptr_t)kgbase);
    __builtin_amdgcn_tensor_load_to_lds(g0, kG1, z4, z4, z8, 0);
  } else if (wave == 1) {
    v4u g0 = tdm_group0((unsigned)(uintptr_t)&Vs[0][0][0],
                        (unsigned long long)(uintptr_t)vgbase);
    __builtin_amdgcn_tensor_load_to_lds(g0, vG1, z4, z4, z8, 0);
  }
#endif

  float mrow[8], lrow[8];
  v8f o[4] = {};
#pragma unroll
  for (int r = 0; r < 8; ++r) { mrow[r] = -3.0e38f; lrow[r] = 0.0f; }

  for (int it = 0; it < ntiles; ++it) {
    const int s0  = it << 5;
    const int cur = it & 1;

#if USE_TDM
    // Tile `it` was DMA'd earlier; drain TENSORcnt in the issuing waves and
    // publish to the workgroup.
    __builtin_amdgcn_s_wait_tensorcnt(0);
    __syncthreads();
    // Overlap: kick off DMA of tile it+1 into the other buffer while all
    // 8 waves run WMMA on tile `it`.
    if (it + 1 < ntiles) {
      const int nxt = (it + 1) & 1;
      if (wave == 0) {
        v4u g0 = tdm_group0((unsigned)(uintptr_t)&Ks[nxt][0][0],
                            (unsigned long long)(uintptr_t)(kgbase + (size_t)(s0 + 32) * DK));
        __builtin_amdgcn_tensor_load_to_lds(g0, kG1, z4, z4, z8, 0);
      } else if (wave == 1) {
        v4u g0 = tdm_group0((unsigned)(uintptr_t)&Vs[nxt][0][0],
                            (unsigned long long)(uintptr_t)(vgbase + s0 + 32));
        __builtin_amdgcn_tensor_load_to_lds(g0, vG1, z4, z4, z8, 0);
      }
    }
#else
    // Fallback: cooperative staging, 16B per thread.
    {
      const int row = tid >> 3, cb = (tid & 7) * 8;
      *reinterpret_cast<U128*>(&Ks[cur][row][cb]) =
          *reinterpret_cast<const U128*>(kgbase + (size_t)(s0 + row) * DK + cb);
      const int d = tid >> 2, cb2 = (tid & 3) * 8;
      *reinterpret_cast<U128*>(&Vs[cur][d][cb2]) =
          *reinterpret_cast<const U128*>(vgbase + (size_t)d * SEQ + s0 + cb2);
    }
    __syncthreads();
#endif

    // scores: batch all 4 K-fragments, then the 4-WMMA chain (K=64 per tile)
    v16h kb[4];
    kb[0] = load_frag_b(&Ks[cur][0][0], 72, lane);     // n=0, k=0..31
    kb[1] = load_frag_b(&Ks[cur][0][32], 72, lane);    // n=0, k=32..63
    kb[2] = load_frag_b(&Ks[cur][16][0], 72, lane);    // n=1, k=0..31
    kb[3] = load_frag_b(&Ks[cur][16][32], 72, lane);   // n=1, k=32..63
    v8f sc[2];
#pragma unroll
    for (int n = 0; n < 2; ++n) {
      v8f s = {};
      s = __builtin_amdgcn_wmma_f32_16x16x32_f16(false, qa0, false, kb[2 * n],
                                                 (short)0, s, false, false);
      s = __builtin_amdgcn_wmma_f32_16x16x32_f16(false, qa1, false, kb[2 * n + 1],
                                                 (short)0, s, false, false);
      const int kpos = s0 + n * 16 + col0;
      if (kpos >= vlen) {
#pragma unroll
        for (int r = 0; r < 8; ++r) s[r] -= 1.0e12f;  // additive mask, matches reference
      }
      sc[n] = s;
    }

    // online softmax: row max via DPP butterfly (pure VALU)
#pragma unroll
    for (int r = 0; r < 8; ++r) {
      const float mx = rowmax16(fmaxf(sc[0][r], sc[1][r]));
      const float mnew = fmaxf(mrow[r], mx);
      const float corr = __expf(mrow[r] - mnew);
      const float p0 = __expf(sc[0][r] - mnew);
      const float p1 = __expf(sc[1][r] - mnew);
      mrow[r] = mnew;
      lrow[r] *= corr;
#pragma unroll
      for (int g = 0; g < 4; ++g) o[g][r] *= corr;
      // re-layout C->A through per-wave LDS patch
      Ps[wave][r + 8 * grp][col0]      = (_Float16)p0;
      Ps[wave][r + 8 * grp][16 + col0] = (_Float16)p1;
    }

    // batch P fragment + all 4 V-fragments, then the WMMA chain
    v16h pa = load_frag_a(&Ps[wave][0][0], 40, lane);
    v16h vb[4];
#pragma unroll
    for (int g = 0; g < 4; ++g) vb[g] = load_frag_b(&Vs[cur][g * 16][0], 40, lane);

    // row sums via WMMA against all-ones B: D[m,n] = sum_k P[m,k] for every n
    {
      v8f rs = {};
      rs = __builtin_amdgcn_wmma_f32_16x16x32_f16(false, pa, false, ones,
                                                  (short)0, rs, false, false);
#pragma unroll
      for (int r = 0; r < 8; ++r) lrow[r] += rs[r];
    }

    // PV: O(16x64) += P(16x32) x V(32x64)
#pragma unroll
    for (int g = 0; g < 4; ++g)
      o[g] = __builtin_amdgcn_wmma_f32_16x16x32_f16(false, pa, false, vb[g],
                                                    (short)0, o[g], false, false);
    __syncthreads();   // everyone done with buf `cur` before it is re-filled
  }

  // epilogue: normalize, apply query mask, write (B, Sq, H*dk) f32
#pragma unroll
  for (int r = 0; r < 8; ++r) {
    const int qpos = q0 + wave * 16 + r + 8 * grp;
    const float inv = 1.0f / lrow[r];
    const bool valid = qpos < qlen;
    float* orow = Out + ((size_t)b * SEQ + qpos) * (HEADS * DK) + h * DK + col0;
#pragma unroll
    for (int g = 0; g < 4; ++g) {
      orow[g * 16] = valid ? o[g][r] * inv : 0.0f;
    }
  }
}

extern "C" void kernel_launch(void* const* d_in, const int* in_sizes, int n_in,
                              void* d_out, int out_size, void* d_ws, size_t ws_size,
                              hipStream_t stream) {
  (void)in_sizes; (void)n_in; (void)out_size; (void)ws_size;
  const float* q  = (const float*)d_in[0];
  const float* k  = (const float*)d_in[1];
  const float* v  = (const float*)d_in[2];
  const float* Wq = (const float*)d_in[3];
  const float* Wk = (const float*)d_in[4];
  const float* Wv = (const float*)d_in[5];
  const int* V_len = (const int*)d_in[6];
  const int* Q_len = (const int*)d_in[7];
  float* out = (float*)d_out;

  const size_t elems = (size_t)4 * HEADS * SEQ * DK;  // 4M halves = 8 MB each
  _Float16* q16  = (_Float16*)d_ws;
  _Float16* k16  = q16 + elems;
  _Float16* v16T = k16 + elems;

  dim3 bp(256);
  dim3 gp(8192 / 128, DMODEL / 64);  // (64, 8)
  proj_gemm_kernel<0><<<gp, bp, 0, stream>>>(q, Wq, q16, 0.125f);  // fold 1/sqrt(dk) into Q
  proj_gemm_kernel<1><<<gp, bp, 0, stream>>>(k, Wk, k16, 1.0f);
  proj_gemm_kernel<2><<<gp, bp, 0, stream>>>(v, Wv, v16T, 1.0f);

  dim3 ga(4 * HEADS, SEQ / 128);     // (32, 16)
  flash_attn_kernel<<<ga, bp, 0, stream>>>(q16, k16, v16T, V_len, Q_len, out);
}
